// SDM_Attention_28063316312661
// MI455X (gfx1250) — matrix operand order repaired and checked by reading, not compiled, not run on hardware
//
#include <hip/hip_runtime.h>
#include <cstdint>
#include <cstddef>

// ---------------------------------------------------------------------------
// MI455X / gfx1250: all GEMM stages via v_wmma_f32_16x16x32_bf16.
// Fragments load as 2 x 16B contiguous chunks per lane. GEMM A-panels are
// DMA'd to LDS by the Tensor Data Mover (tensor_load_to_lds + s_wait_tensorcnt)
// where the panel fits; B streams from global with prefetch. fp32->bf16 uses
// native v_cvt_pk_bf16_f32 where available.
// ---------------------------------------------------------------------------

typedef __attribute__((ext_vector_type(16))) __bf16        v16bf;
typedef __attribute__((ext_vector_type(2)))  __bf16        v2bf;
typedef __attribute__((ext_vector_type(8)))  float         v8f;
typedef __attribute__((ext_vector_type(4)))  unsigned int  u32x4;

#define BN   16
#define HWN  4096
#define QN   100
#define QP   112   // Q padded to 16
#define QP2  128   // Q padded to 32 (WMMA K chunks)
#define CN   256
#define HN   8
#define DN   32
#define HIDN 2048
#define SCALEF 0.17677669529663687f  // 1/sqrt(32)

#if defined(__has_builtin)
#if __has_builtin(__builtin_amdgcn_tensor_load_to_lds) && \
    __has_builtin(__builtin_amdgcn_s_wait_tensorcnt)
#define HAVE_TDM 1
#endif
#if __has_builtin(__builtin_amdgcn_cvt_pk_bf16_f32)
#define HAVE_PKCVT 1
#endif
#endif
#ifndef HAVE_TDM
#define HAVE_TDM 0
#endif
#ifndef HAVE_PKCVT
#define HAVE_PKCVT 0
#endif

__device__ __forceinline__ int lane() { return threadIdx.x & 31; }

__device__ __forceinline__ float b2f(__bf16 v) {
  unsigned int w = ((unsigned int)__builtin_bit_cast(unsigned short, v)) << 16;
  return __builtin_bit_cast(float, w);
}
__device__ __forceinline__ __bf16 f2b(float f) {
  unsigned int w = __builtin_bit_cast(unsigned int, f);
  unsigned short r = (unsigned short)((w + 0x7FFFu + ((w >> 16) & 1u)) >> 16);
  return __builtin_bit_cast(__bf16, r);
}
__device__ __forceinline__ v2bf f2b_pk(float a, float b) {
#if HAVE_PKCVT
  return __builtin_amdgcn_cvt_pk_bf16_f32(a, b);
#else
  v2bf r; r[0] = f2b(a); r[1] = f2b(b); return r;
#endif
}

// A-operand elem e -> K = (e<8?0:8) + hi*8 + e  (two contiguous 8-elem runs)
__device__ __forceinline__ int afk(int e, int hi) {
  return ((e >> 3) << 3) + hi * 8 + e;
}

union F16  { u32x4 u[2]; __bf16 hh[16]; v16bf v; };
union F8   { u32x4 u;    __bf16 hh[8]; };
union PF16 { v2bf p2[8]; v16bf v; };
union P4   { v2bf p2[4]; u32x4 u; };

__device__ __forceinline__ v16bf load_a_frag_g(const __bf16* row) {
  int hi = lane() >> 4;
  F16 x;
  x.u[0] = *(const u32x4*)(row + hi * 8);
  x.u[1] = *(const u32x4*)(row + 16 + hi * 8);
  return x.v;
}
__device__ __forceinline__ v16bf load_b_frag_g(const __bf16* row) {
  int koff = (lane() >> 4) * 8;
  F16 x;
  x.u[0] = *(const u32x4*)(row + koff);
  x.u[1] = *(const u32x4*)(row + koff + 16);
  return x.v;
}
__device__ __forceinline__ v8f wmma_bf16(v16bf a, v16bf b, v8f c) {
  return __builtin_amdgcn_wmma_f32_16x16x32_bf16(false, a, false, b,
                                                 (short)0, c, false, false);
}

#if HAVE_TDM
typedef __attribute__((ext_vector_type(8))) int s32x8;
typedef __attribute__((ext_vector_type(4))) int s32x4;
// TDM: DMA a [16 rows x K elems] bf16 panel (row stride = K) into LDS.
__device__ __forceinline__ void tdm_load_a16(const __bf16* gsrc,
                                             unsigned lds_off, int K) {
  unsigned long ga = (unsigned long)(uintptr_t)gsrc;
  unsigned kd = (unsigned)K;
  u32x4 g0 = {1u, lds_off, (unsigned)ga,
              (unsigned)((ga >> 32) & 0x01ffffffu) | 0x80000000u};
  s32x8 g1 = {0, 0, 0, 0, 0, 0, 0, 0};
  g1[0] = (int)(1u << 16);                                   // data_size = 2B
  g1[1] = (int)((kd & 0xffffu) << 16);                       // tensor_dim0 lo
  g1[2] = (int)(((kd >> 16) & 0xffffu) | (16u << 16));       // dim0 hi, dim1=16
  g1[3] = (int)((kd & 0xffffu) << 16);                       // tile_dim0 = K
  g1[4] = 16;                                                // tile_dim1 = 16
  g1[5] = (int)kd;                                           // dim0_stride = K
  s32x4 z4 = {0, 0, 0, 0};
#if __clang_major__ >= 23
  s32x8 z8 = {0, 0, 0, 0, 0, 0, 0, 0};
  __builtin_amdgcn_tensor_load_to_lds(g0, g1, z4, z4, z8, 0);
#else
  __builtin_amdgcn_tensor_load_to_lds(g0, g1, z4, z4, 0);
#endif
}
#endif

// ---------------------------------------------------------------------------
__global__ void k_cvt_bf16(const float* __restrict__ s, __bf16* __restrict__ d,
                           long n) {
  long i = ((long)blockIdx.x * blockDim.x + threadIdx.x) * 2;
  if (i < n) *(v2bf*)(d + i) = f2b_pk(s[i], s[i + 1]);
}

// fp32 W[K,N] -> bf16 W^T[N,K], 32x32 LDS tiles
__global__ __launch_bounds__(256)
void k_cvt_t(const float* __restrict__ S, __bf16* __restrict__ D, int K, int N) {
  __shared__ __bf16 tile[32][33];
  int k0 = blockIdx.y * 32, n0 = blockIdx.x * 32;
  int t = threadIdx.x;
  for (int i = t; i < 512; i += 256) {
    int r = i >> 4, c = (i & 15) * 2;
    const float* sp = S + (long)(k0 + r) * N + n0 + c;
    v2bf pr = f2b_pk(sp[0], sp[1]);
    tile[r][c] = pr[0];
    tile[r][c + 1] = pr[1];
  }
  __syncthreads();
  for (int i = t; i < 1024; i += 256) {
    int r = i >> 5, c = i & 31;
    D[(long)(n0 + r) * K + k0 + c] = tile[c][r];
  }
}

// bf16 head-split transpose: S[b, r, CN] (head h slice) -> D[bh, d, rpad]
__global__ __launch_bounds__(256)
void k_thd(const __bf16* __restrict__ S, __bf16* __restrict__ D,
           int rows, int rpad) {
  __shared__ __bf16 tile[32][33];
  int bh = blockIdx.y, b = bh >> 3, h = bh & 7;
  int r0 = blockIdx.x * 32, t = threadIdx.x;
  for (int i = t; i < 1024; i += 256) {
    int r = i >> 5, c = i & 31;
    int rr = r0 + r;
    tile[r][c] = (rr < rows) ? S[((long)b * rows + rr) * CN + h * DN + c]
                             : f2b(0.f);
  }
  __syncthreads();
  for (int i = t; i < 1024; i += 256) {
    int d = i >> 5, c = i & 31;
    D[((long)bh * DN + d) * rpad + r0 + c] = tile[c][d];
  }
}

// Generic bf16 GEMM: out = act(A[M,K] @ BT[N,K]^T + bias).
// STAGE=1: A panel DMA'd to LDS via TDM (4-wave reuse); STAGE=0: direct global.
template <int STAGE>
__global__ __launch_bounds__(128)
void k_gemm_bf16(const __bf16* __restrict__ A, const __bf16* __restrict__ BT,
                 const float* __restrict__ bias, __bf16* __restrict__ outB,
                 float* __restrict__ outF, int M, int N, int K,
                 int relu, int group, int keep) {
  int t = threadIdx.x, w = t >> 5, l = lane();
  int m = l & 15, hi = l >> 4;
  int m0 = blockIdx.y * 16, n0 = blockIdx.x * 64;
  const __bf16* brow = BT + (long)(n0 + w * 16 + m) * K;
#if HAVE_TDM
  extern __shared__ __bf16 AsT[];
  const __bf16* arow;
  if (STAGE) {
    if (w == 0) {
      tdm_load_a16(A + (long)m0 * K, (unsigned)(uintptr_t)AsT, K);
      __builtin_amdgcn_s_wait_tensorcnt(0);
    }
    __syncthreads();
    arow = AsT + m * K;
  } else {
    arow = A + (long)(m0 + m) * K;
  }
#else
  const __bf16* arow = A + (long)(m0 + m) * K;
#endif
  v8f acc0 = {}, acc1 = {};
  for (int k0 = 0; k0 < K; k0 += 64) {
    __builtin_prefetch(brow + k0 + 256, 0, 1);
    acc0 = wmma_bf16(load_a_frag_g(arow + k0), load_b_frag_g(brow + k0), acc0);
    acc1 = wmma_bf16(load_a_frag_g(arow + k0 + 32),
                     load_b_frag_g(brow + k0 + 32), acc1);
  }
  v8f acc = acc0 + acc1;
  int col = n0 + w * 16 + m;
  float bv = bias ? bias[col] : 0.f;
#pragma unroll
  for (int j = 0; j < 8; j += 2) {
    int r0_ = m0 + j + hi * 8, r1_ = r0_ + 1;
    float v0 = acc[j] + bv, v1 = acc[j + 1] + bv;
    if (relu) { v0 = fmaxf(v0, 0.f); v1 = fmaxf(v1, 0.f); }
    v2bf pr = f2b_pk(v0, v1);
    if (outB) {
      outB[(long)r0_ * N + col] = pr[0];
      outB[(long)r1_ * N + col] = pr[1];
    }
    if (outF) {
      int g0_ = r0_ % group, g1_ = r1_ % group;
      if (g0_ < keep) outF[((long)(r0_ / group) * keep + g0_) * N + col] = v0;
      if (g1_ < keep) outF[((long)(r1_ / group) * keep + g1_) * N + col] = v1;
    }
  }
}

// A[b,h,q,hw] = SCALE * l_k @ v_k^T. LDS-free: K-dim is d, contiguous in both.
__global__ __launch_bounds__(128)
void k_scores(const __bf16* __restrict__ lk, const __bf16* __restrict__ vk,
              __bf16* __restrict__ Ab) {
  int t = threadIdx.x, w = t >> 5, l = lane();
  int ktg = blockIdx.x, qt = blockIdx.y, bh = blockIdx.z;
  int b = bh >> 3, h = bh & 7;
  int m = l & 15, hi = l >> 4;
  const __bf16* arow = lk + ((long)b * QN + qt * 16 + m) * CN + h * DN;
  int kt = ktg * 4 + w;
  const __bf16* brow = vk + ((long)b * HWN + kt * 16 + m) * CN + h * DN;
  v8f acc = {};
  acc = wmma_bf16(load_a_frag_g(arow), load_b_frag_g(brow), acc);
  long base = (long)bh * QP2 * HWN;
#pragma unroll
  for (int j = 0; j < 8; j += 2) {
    int q0 = qt * 16 + j + hi * 8;
    v2bf pr = f2b_pk(acc[j] * SCALEF, acc[j + 1] * SCALEF);
    Ab[base + (long)q0 * HWN + kt * 16 + m] = pr[0];
    Ab[base + (long)(q0 + 1) * HWN + kt * 16 + m] = pr[1];
  }
}

// softmax stats over HW (one wave per q row), b128 loads
__global__ __launch_bounds__(128)
void k_rowstats(const __bf16* __restrict__ Ab, float* __restrict__ rowM,
                float* __restrict__ rowS) {
  int w = threadIdx.x >> 5, l = lane();
  int q = blockIdx.x * 4 + w;
  int bh = blockIdx.y;
  if (q >= QN) {
    if (l == 0) { rowM[bh * QP + q] = 0.f; rowS[bh * QP + q] = 1.f; }
    return;
  }
  const __bf16* Ar = Ab + ((long)bh * QP2 + q) * HWN;
  float m = -1e30f;
  for (int i = l * 8; i < HWN; i += 256) {
    F8 x; x.u = *(const u32x4*)(Ar + i);
#pragma unroll
    for (int e = 0; e < 8; ++e) m = fmaxf(m, b2f(x.hh[e]));
  }
#pragma unroll
  for (int off = 16; off > 0; off >>= 1) m = fmaxf(m, __shfl_xor(m, off, 32));
  float s = 0.f;
  for (int i = l * 8; i < HWN; i += 256) {
    F8 x; x.u = *(const u32x4*)(Ar + i);
#pragma unroll
    for (int e = 0; e < 8; ++e) s += __expf(b2f(x.hh[e]) - m);
  }
#pragma unroll
  for (int off = 16; off > 0; off >>= 1) s += __shfl_xor(s, off, 32);
  if (l == 0) { rowM[bh * QP + q] = m; rowS[bh * QP + q] = s; }
}

// softmax stats over Q (thread per hw column; coalesced across threads)
__global__ __launch_bounds__(256)
void k_colstats(const __bf16* __restrict__ Ab, float* __restrict__ colM,
                float* __restrict__ colS) {
  int hw = blockIdx.x * 256 + threadIdx.x;
  int bh = blockIdx.y;
  const __bf16* Ac = Ab + (long)bh * QP2 * HWN + hw;
  float m = -1e30f;
  for (int q = 0; q < QN; ++q) m = fmaxf(m, b2f(Ac[(long)q * HWN]));
  float s = 0.f;
  for (int q = 0; q < QN; ++q) s += __expf(b2f(Ac[(long)q * HWN]) - m);
  colM[bh * HWN + hw] = m;
  colS[bh * HWN + hw] = s;
}

// v_a^T = v_v^T @ P^T. A-op from vvT (hw-contiguous), B-op = exp(A row)
// with per-lane-uniform stats. 2 waves/block = two d-tiles. LDS-free.
__global__ __launch_bounds__(64)
void k_va(const __bf16* __restrict__ Ab, const float* __restrict__ rowM,
          const float* __restrict__ rowS, const __bf16* __restrict__ vvT,
          __bf16* __restrict__ va) {
  int t = threadIdx.x, w = t >> 5, l = lane();
  int qt = blockIdx.x, bh = blockIdx.y;
  int m = l & 15, hi = l >> 4, koff = hi * 8;
  int q = qt * 16 + m;
  float rm = rowM[bh * QP + q];
  float ri = 1.f / rowS[bh * QP + q];
  const __bf16* arow = vvT + ((long)bh * DN + w * 16 + m) * HWN;
  const __bf16* brow = Ab + ((long)bh * QP2 + q) * HWN;
  v8f acc0 = {}, acc1 = {};
  for (int k0 = 0; k0 < HWN; k0 += 64) {
#pragma unroll
    for (int s = 0; s < 2; ++s) {
      int kk = k0 + s * 32;
      v16bf af = load_a_frag_g(arow + kk);
      F16 x;
      x.u[0] = *(const u32x4*)(brow + kk + koff);
      x.u[1] = *(const u32x4*)(brow + kk + koff + 16);
      PF16 pf;
#pragma unroll
      for (int e = 0; e < 16; e += 2)
        pf.p2[e >> 1] = f2b_pk(__expf(b2f(x.hh[e]) - rm) * ri,
                               __expf(b2f(x.hh[e + 1]) - rm) * ri);
      if (s == 0) acc0 = wmma_bf16(af, pf.v, acc0);
      else        acc1 = wmma_bf16(af, pf.v, acc1);
    }
  }
  v8f acc = acc0 + acc1;
  P4 o;
#pragma unroll
  for (int j = 0; j < 8; j += 2) o.p2[j >> 1] = f2b_pk(acc[j], acc[j + 1]);
  *(u32x4*)&va[((long)bh * QP + q) * DN + w * 16 + hi * 8] = o.u;
}

// l_a = P_col^T @ l_v. Fused per-tile LDS transpose of A; B-op from lvT.
__global__ __launch_bounds__(128)
void k_la(const __bf16* __restrict__ Ab, const float* __restrict__ colM,
          const float* __restrict__ colS, const __bf16* __restrict__ lvT,
          __bf16* __restrict__ la) {
  __shared__ alignas(16) __bf16 AtT[4][16][40];
  int t = threadIdx.x, w = t >> 5, l = lane();
  int bh = blockIdx.y;
  int hw0 = blockIdx.x * 64 + w * 16;
  int m = l & 15, hi = l >> 4;
  int hw = hw0 + m;
  float cm = colM[bh * HWN + hw];
  float ci = 1.f / colS[bh * HWN + hw];
  v8f acc0 = {}, acc1 = {};
  for (int ks = 0; ks < 4; ++ks) {
    {
      const __bf16* src = Ab + ((long)bh * QP2 + ks * 32 + l) * HWN + hw0;
      F16 x;
      x.u[0] = *(const u32x4*)src;
      x.u[1] = *(const u32x4*)(src + 8);
#pragma unroll
      for (int n = 0; n < 16; ++n) AtT[w][n][l] = x.hh[n];
    }
    __syncthreads();
    F16 a;
    a.u[0] = *(const u32x4*)&AtT[w][m][hi * 8];
    a.u[1] = *(const u32x4*)&AtT[w][m][16 + hi * 8];
    PF16 pf;
#pragma unroll
    for (int e = 0; e < 16; e += 2) {
      int qg0 = ks * 32 + afk(e, hi);
      int qg1 = ks * 32 + afk(e + 1, hi);
      float p0 = (qg0 < QN) ? __expf(b2f(a.hh[e]) - cm) * ci : 0.f;
      float p1 = (qg1 < QN) ? __expf(b2f(a.hh[e + 1]) - cm) * ci : 0.f;
      pf.p2[e >> 1] = f2b_pk(p0, p1);
    }
    const __bf16* b0row = lvT + ((long)bh * DN + m) * QP2 + ks * 32;
    const __bf16* b1row = lvT + ((long)bh * DN + 16 + m) * QP2 + ks * 32;
    acc0 = wmma_bf16(pf.v, load_b_frag_g(b0row), acc0);
    acc1 = wmma_bf16(pf.v, load_b_frag_g(b1row), acc1);
    __syncthreads();
  }
#pragma unroll
  for (int j = 0; j < 8; j += 2) {
    long o0 = ((long)bh * HWN + hw0 + j + hi * 8) * DN;
    long o1 = o0 + DN;
    v2bf pa = f2b_pk(acc0[j], acc0[j + 1]);
    v2bf pb = f2b_pk(acc1[j], acc1[j + 1]);
    la[o0 + m] = pa[0];      la[o1 + m] = pa[1];
    la[o0 + 16 + m] = pb[0]; la[o1 + 16 + m] = pb[1];
  }
}

// fm = mean_h(v_a @ l_a^T): attn_mask fp32 to d_out + bf16 copy for MLP
__global__ __launch_bounds__(128)
void k_fmul(const __bf16* __restrict__ va, const __bf16* __restrict__ la,
            float* __restrict__ out_base, __bf16* __restrict__ fmb) {
  int t = threadIdx.x, w = t >> 5, l = lane();
  int b = blockIdx.z, qt = blockIdx.y;
  int hw0 = (blockIdx.x * 4 + w) * 16;
  int m = l & 15, hi = l >> 4, koff = hi * 8;
  v8f acc = {};
  for (int h = 0; h < HN; ++h) {
    int bh = b * HN + h;
    const __bf16* ap = va + ((long)bh * QP + qt * 16 + m) * DN;
    const __bf16* bp = la + ((long)bh * HWN + hw0 + m) * DN + koff;
    F16 au, bu;
    au.u[0] = *(const u32x4*)(ap + hi * 8);
    au.u[1] = *(const u32x4*)(ap + 16 + hi * 8);
    bu.u[0] = *(const u32x4*)bp;
    bu.u[1] = *(const u32x4*)(bp + 16);
    acc = wmma_bf16(au.v, bu.v, acc);
  }
  int hw = hw0 + m;
#pragma unroll
  for (int j = 0; j < 8; j += 2) {
    int q0 = qt * 16 + j + hi * 8, q1 = q0 + 1;
    float v0 = acc[j] * 0.125f, v1 = acc[j + 1] * 0.125f;
    float s0 = (q0 < QN) ? v0 : 0.f, s1 = (q1 < QN) ? v1 : 0.f;
    v2bf pr = f2b_pk(s0, s1);
    fmb[((long)b * QP + q0) * HWN + hw] = pr[0];
    fmb[((long)b * QP + q1) * HWN + hw] = pr[1];
    if (q0 < QN)
      out_base[(long)BN * QN * CN + ((long)(b * QN + q0)) * HWN + hw] = v0;
    if (q1 < QN)
      out_base[(long)BN * QN * CN + ((long)(b * QN + q1)) * HWN + hw] = v1;
  }
}

// ---------------------------------------------------------------------------
extern "C" void kernel_launch(void* const* d_in, const int* in_sizes, int n_in,
                              void* d_out, int out_size, void* d_ws, size_t ws_size,
                              hipStream_t stream) {
  (void)in_sizes; (void)n_in; (void)out_size; (void)ws_size;
  const float* vis  = (const float*)d_in[0];
  const float* lang = (const float*)d_in[1];
  const float* Wvk  = (const float*)d_in[2];  const float* bvk = (const float*)d_in[3];
  const float* Wvv  = (const float*)d_in[4];  const float* bvv = (const float*)d_in[5];
  const float* Wlk  = (const float*)d_in[6];  const float* blk = (const float*)d_in[7];
  const float* Wlv  = (const float*)d_in[8];  const float* blv = (const float*)d_in[9];
  const float* W0   = (const float*)d_in[10]; const float* b0  = (const float*)d_in[11];
  const float* W1   = (const float*)d_in[12]; const float* b1  = (const float*)d_in[13];

  char* p = (char*)d_ws;
  auto alloc = [&](size_t elems, size_t esz) -> void* {
    void* r = (void*)p;
    p += ((elems * esz) + 255) & ~(size_t)255;
    return r;
  };
  __bf16* visb = (__bf16*)alloc((size_t)BN * HWN * CN, 2);
  __bf16* langb= (__bf16*)alloc((size_t)BN * QN * CN, 2);
  __bf16* wvkT = (__bf16*)alloc((size_t)CN * CN, 2);
  __bf16* wvvT = (__bf16*)alloc((size_t)CN * CN, 2);
  __bf16* wlkT = (__bf16*)alloc((size_t)CN * CN, 2);
  __bf16* wlvT = (__bf16*)alloc((size_t)CN * CN, 2);
  __bf16* w0T  = (__bf16*)alloc((size_t)HWN * HIDN, 2);
  __bf16* w1T  = (__bf16*)alloc((size_t)HIDN * CN, 2);
  __bf16* vkb  = (__bf16*)alloc((size_t)BN * HWN * CN, 2);
  __bf16* vvb  = (__bf16*)alloc((size_t)BN * HWN * CN, 2);
  __bf16* lkb  = (__bf16*)alloc((size_t)BN * QN * CN, 2);
  __bf16* lvb  = (__bf16*)alloc((size_t)BN * QN * CN, 2);
  __bf16* vvT  = (__bf16*)alloc((size_t)BN * HN * DN * HWN, 2);
  __bf16* lvT  = (__bf16*)alloc((size_t)BN * HN * DN * QP2, 2);
  __bf16* Abuf = (__bf16*)alloc((size_t)BN * HN * QP2 * HWN, 2);
  __bf16* vab  = (__bf16*)alloc((size_t)BN * HN * QP * DN, 2);
  __bf16* lab  = (__bf16*)alloc((size_t)BN * HN * HWN * DN, 2);
  __bf16* fmb  = (__bf16*)alloc((size_t)BN * QP * HWN, 2);
  __bf16* hidb = (__bf16*)alloc((size_t)BN * QP * HIDN, 2);
  float*  rowM = (float*)alloc((size_t)BN * HN * QP, 4);
  float*  rowS = (float*)alloc((size_t)BN * HN * QP, 4);
  float*  colM = (float*)alloc((size_t)BN * HN * HWN, 4);
  float*  colS = (float*)alloc((size_t)BN * HN * HWN, 4);

  auto cvt = [&](const float* s, __bf16* d, long n) {
    k_cvt_bf16<<<dim3((unsigned)((n / 2 + 255) / 256)), 256, 0, stream>>>(s, d, n);
  };
  cvt(vis,  visb,  (long)BN * HWN * CN);
  cvt(lang, langb, (long)BN * QN * CN);
  k_cvt_t<<<dim3(8, 8),    256, 0, stream>>>(Wvk, wvkT, CN, CN);
  k_cvt_t<<<dim3(8, 8),    256, 0, stream>>>(Wvv, wvvT, CN, CN);
  k_cvt_t<<<dim3(8, 8),    256, 0, stream>>>(Wlk, wlkT, CN, CN);
  k_cvt_t<<<dim3(8, 8),    256, 0, stream>>>(Wlv, wlvT, CN, CN);
  k_cvt_t<<<dim3(64, 128), 256, 0, stream>>>(W0, w0T, HWN, HIDN);
  k_cvt_t<<<dim3(8, 64),   256, 0, stream>>>(W1, w1T, HIDN, CN);

  // projections (K=256): TDM-staged A panel (8KB LDS)
  size_t shp = (size_t)16 * CN * 2;
  k_gemm_bf16<1><<<dim3(4, 4096), 128, shp, stream>>>(visb, wvkT, bvk, vkb,
      nullptr, BN * HWN, CN, CN, 0, 1, 1);
  k_gemm_bf16<1><<<dim3(4, 4096), 128, shp, stream>>>(visb, wvvT, bvv, vvb,
      nullptr, BN * HWN, CN, CN, 0, 1, 1);
  k_gemm_bf16<1><<<dim3(4, 100), 128, shp, stream>>>(langb, wlkT, blk, lkb,
      nullptr, BN * QN, CN, CN, 0, 1, 1);
  k_gemm_bf16<1><<<dim3(4, 100), 128, shp, stream>>>(langb, wlvT, blv, lvb,
      nullptr, BN * QN, CN, CN, 0, 1, 1);

  k_thd<<<dim3(HWN / 32, BN * HN), 256, 0, stream>>>(vvb, vvT, HWN, HWN);
  k_thd<<<dim3(QP2 / 32, BN * HN), 256, 0, stream>>>(lvb, lvT, QN, QP2);

  k_scores<<<dim3(64, 7, BN * HN), 128, 0, stream>>>(lkb, vkb, Abuf);
  k_rowstats<<<dim3(28, BN * HN), 128, 0, stream>>>(Abuf, rowM, rowS);
  k_colstats<<<dim3(16, BN * HN), 256, 0, stream>>>(Abuf, colM, colS);
  k_va<<<dim3(7, BN * HN), 64, 0, stream>>>(Abuf, rowM, rowS, vvT, vab);
  k_la<<<dim3(64, BN * HN), 128, 0, stream>>>(Abuf, colM, colS, lvT, lab);
  k_fmul<<<dim3(64, 7, BN), 128, 0, stream>>>(vab, lab, (float*)d_out, fmb);

  // MLP1 (K=4096): A panel too big for LDS -> direct-global pipeline
  k_gemm_bf16<0><<<dim3(HIDN / 64, (BN * QP) / 16), 128, 0, stream>>>(
      fmb, w0T, b0, hidb, nullptr, BN * QP, HIDN, HWN, 1, 1, 1);
  // MLP2 (K=2048): TDM-staged A panel (64KB LDS)
  k_gemm_bf16<1><<<dim3(CN / 64, (BN * QP) / 16), 128,
                   (size_t)16 * HIDN * 2, stream>>>(
      hidb, w1T, b1, nullptr, (float*)d_out, BN * QP, CN, HIDN, 0, QP, QN);
}
